// TransformerEncoder_9955734192178
// MI455X (gfx1250) — matrix operand order repaired and checked by reading, not compiled, not run on hardware
//
#include <hip/hip_runtime.h>

#define DEV __device__ __forceinline__

typedef __bf16 v16bf __attribute__((ext_vector_type(16)));
typedef float  v8f   __attribute__((ext_vector_type(8)));

// ---------- bf16 helpers (round-to-nearest-even) ----------
DEV unsigned short f32_to_bf16(float f){
  union { float f; unsigned int u; } x; x.f = f;
  unsigned int r = 0x7FFFu + ((x.u >> 16) & 1u);
  return (unsigned short)((x.u + r) >> 16);
}
DEV __bf16 bf_bits(unsigned short s){
  union { unsigned short u; __bf16 b; } x; x.u = s; return x.b;
}

DEV v8f wmma_bf16(v16bf a, v16bf b, v8f c){
  // 8 args: (neg_a, A, neg_b, B, c_mod, C, reuse_a, reuse_b)
  return __builtin_amdgcn_wmma_f32_16x16x32_bf16(false, a, false, b, (short)0, c, false, false);
}

// ---------- CDNA5 async memory->LDS (ASYNCcnt-tracked) ----------
DEV void async_copy_b128_to_lds(const unsigned short* gsrc, unsigned short* lds_dst){
  // LDS aperture: generic-pointer low 32 bits are the LDS byte address (ISA 10.2)
  unsigned lds_addr = (unsigned)(uintptr_t)lds_dst;
  unsigned long long gaddr = (unsigned long long)(uintptr_t)gsrc;
  asm volatile("global_load_async_to_lds_b128 %0, %1, off"
               :: "v"(lds_addr), "v"(gaddr) : "memory");
}

// ---------- WMMA fragment loaders (CDNA5 wave32 layouts) ----------
// A fragment: 16x32 bf16 tile from row-major A, origin (row0, k0).
// Lanes 0-15: M=lane, K {0..7,16..23}; lanes 16-31: M=lane-16, K {8..15,24..31}.
DEV v16bf load_a_frag(const unsigned short* A, int lda, int row0, int k0){
  int lane = threadIdx.x & 31;
  const unsigned short* p = A + (size_t)(row0 + (lane & 15)) * lda + k0 + ((lane >> 4) << 3);
  v16bf a;
#pragma unroll
  for (int j = 0; j < 8; ++j) a[j]     = bf_bits(p[j]);
#pragma unroll
  for (int j = 0; j < 8; ++j) a[8 + j] = bf_bits(p[16 + j]);
  return a;
}

// B fragment: 32x16 bf16 tile from row-major B, origin (k0, col0).
// N = lane%16, K {0..7,16..23} for lanes 0-15, {8..15,24..31} for lanes 16-31.
DEV v16bf load_b_frag(const unsigned short* B, int ldb, int k0, int col0){
  int lane = threadIdx.x & 31;
  const unsigned short* p = B + (size_t)(k0 + ((lane >> 4) << 3)) * ldb + col0 + (lane & 15);
  v16bf b;
#pragma unroll
  for (int j = 0; j < 8; ++j) b[j] = bf_bits(p[(size_t)j * ldb]);
#pragma unroll
  for (int j = 0; j < 8; ++j) b[8 + j] = bf_bits(p[(size_t)(16 + j) * ldb]);
  return b;
}

// B fragment from a transposed source: B(k,n) = M[n0+n][k0+k]  (for Q·K^T).
DEV v16bf load_bT_frag(const unsigned short* M, int ld, int n0, int k0){
  int lane = threadIdx.x & 31;
  const unsigned short* p = M + (size_t)(n0 + (lane & 15)) * ld + k0 + ((lane >> 4) << 3);
  v16bf b;
#pragma unroll
  for (int j = 0; j < 8; ++j) b[j] = bf_bits(p[j]);
#pragma unroll
  for (int j = 0; j < 8; ++j) b[8 + j] = bf_bits(p[16 + j]);
  return b;
}

// ---------- elementwise kernels ----------
__global__ void prep_kernel(const float* __restrict__ x, const float* __restrict__ timing,
                            float* __restrict__ h, unsigned short* __restrict__ hb, int total){
  int i = blockIdx.x * blockDim.x + threadIdx.x;
  if (i >= total) return;
  float v = x[i] + timing[i & (512 * 1024 - 1)];
  h[i] = v;
  hb[i] = f32_to_bf16(v);
}

__global__ void cast_kernel(const float* __restrict__ src, unsigned short* __restrict__ dst, int n){
  int i = blockIdx.x * blockDim.x + threadIdx.x;
  if (i < n) dst[i] = f32_to_bf16(src[i]);
}

// ---------- QKV projection: per-head 16x64 tiles, K-loop over 1024 ----------
__global__ __launch_bounds__(128) void qkv_kernel(
    const unsigned short* __restrict__ hb,
    const unsigned short* __restrict__ wq,
    const unsigned short* __restrict__ wk,
    const unsigned short* __restrict__ wv,
    unsigned short* __restrict__ qout,
    unsigned short* __restrict__ kout,
    unsigned short* __restrict__ vout)
{
  const int DM = 1024, DQ = 64, S = 512, NH = 16;
  __shared__ unsigned short As[16 * 32];
  int tid = threadIdx.x, wave = tid >> 5, lane = tid & 31;
  int r0 = blockIdx.x * 16;
  int head = blockIdx.y;
  const unsigned short* W = (blockIdx.z == 0 ? wq : blockIdx.z == 1 ? wk : wv) + (size_t)head * DM * DQ;
  unsigned short* out = (blockIdx.z == 0 ? qout : blockIdx.z == 1 ? kout : vout);
  v8f acc = {};
  for (int k0 = 0; k0 < DM; k0 += 32){
#pragma unroll
    for (int i = 0; i < 4; ++i){
      int e = tid * 4 + i;                         // 16x32 tile of activations -> LDS
      As[e] = hb[(size_t)(r0 + (e >> 5)) * DM + k0 + (e & 31)];
    }
    if (k0 + 32 < DM)
      __builtin_prefetch(&W[(size_t)(k0 + 32 + lane) * DQ + wave * 16], 0, 1);
    __syncthreads();
    v16bf af = load_a_frag(As, 32, 0, 0);
    v16bf bf = load_b_frag(W, DQ, k0, wave * 16);  // wave owns 16 of 64 head-dims
    acc = wmma_bf16(af, bf, acc);
    __syncthreads();
  }
  int col = wave * 16 + (lane & 15);
  int mb = ((lane >> 4) << 3);
#pragma unroll
  for (int v = 0; v < 8; ++v){
    int r = r0 + mb + v;
    int b = r >> 9, s = r & 511;
    out[((size_t)(b * NH + head) * S + s) * DQ + col] = f32_to_bf16(acc[v]);
  }
}

// ---------- fused flash attention: one wave per 16-row query tile ----------
__global__ __launch_bounds__(256) void attn_kernel(
    const unsigned short* __restrict__ Q,
    const unsigned short* __restrict__ K,
    const unsigned short* __restrict__ V,
    const int* __restrict__ mask,
    unsigned short* __restrict__ O)
{
  const int S = 512, DQ = 64, NH = 16, DM = 1024;
  const float scale = 0.125f;                      // 1/sqrt(64)
  __shared__ unsigned short pbuf[8][16 * 32];      // per-wave P tile (C-layout -> A-layout shuffle)
  int wave = threadIdx.x >> 5, lane = threadIdx.x & 31;
  int head = blockIdx.y, b = blockIdx.z;
  int q0 = (blockIdx.x * 8 + wave) * 16;
  size_t base = (size_t)(b * NH + head) * S * DQ;
  const unsigned short* Qp = Q + base;
  const unsigned short* Kp = K + base;
  const unsigned short* Vp = V + base;
  v16bf qa0 = load_a_frag(Qp, DQ, q0, 0);
  v16bf qa1 = load_a_frag(Qp, DQ, q0, 32);
  v8f o0 = {}, o1 = {}, o2 = {}, o3 = {};
  float mrow[8], lrow[8];
#pragma unroll
  for (int v = 0; v < 8; ++v){ mrow[v] = -3.0e38f; lrow[v] = 0.f; }
  int col = lane & 15;
  int mb = ((lane >> 4) << 3);
  unsigned short* pb = pbuf[wave];

  for (int j0 = 0; j0 < S; j0 += 32){
    // scores S[16x32] = Q(16x64) . K^T, two 16x16 C tiles, contraction 64 = 2 WMMA each
    v8f c0 = {}, c1 = {};
    c0 = wmma_bf16(qa0, load_bT_frag(Kp, DQ, j0,      0),  c0);
    c0 = wmma_bf16(qa1, load_bT_frag(Kp, DQ, j0,      32), c0);
    c1 = wmma_bf16(qa0, load_bT_frag(Kp, DQ, j0 + 16, 0),  c1);
    c1 = wmma_bf16(qa1, load_bT_frag(Kp, DQ, j0 + 16, 32), c1);
    float bias0 = (mask[b * S + j0 + col] == 0)      ? -1e9f : 0.f;
    float bias1 = (mask[b * S + j0 + 16 + col] == 0) ? -1e9f : 0.f;
    float alpha[8];
#pragma unroll
    for (int v = 0; v < 8; ++v){
      float s0 = c0[v] * scale + bias0;
      float s1 = c1[v] * scale + bias1;
      float mx = fmaxf(s0, s1);
#pragma unroll
      for (int d = 1; d < 16; d <<= 1) mx = fmaxf(mx, __shfl_xor(mx, d, 32));
      float mnew = fmaxf(mrow[v], mx);
      float a  = __expf(mrow[v] - mnew);
      float e0 = __expf(s0 - mnew);
      float e1 = __expf(s1 - mnew);
      float rs = e0 + e1;
#pragma unroll
      for (int d = 1; d < 16; d <<= 1) rs += __shfl_xor(rs, d, 32);
      lrow[v] = lrow[v] * a + rs;
      mrow[v] = mnew;
      alpha[v] = a;
      pb[(mb + v) * 32 + col]      = f32_to_bf16(e0);   // P -> LDS, row-major 16x32
      pb[(mb + v) * 32 + 16 + col] = f32_to_bf16(e1);
    }
#pragma unroll
    for (int v = 0; v < 8; ++v){ o0[v] *= alpha[v]; o1[v] *= alpha[v]; o2[v] *= alpha[v]; o3[v] *= alpha[v]; }
    v16bf pa = load_a_frag(pb, 32, 0, 0);               // same-wave LDS ops are in-order
    o0 = wmma_bf16(pa, load_b_frag(Vp, DQ, j0, 0),  o0);
    o1 = wmma_bf16(pa, load_b_frag(Vp, DQ, j0, 16), o1);
    o2 = wmma_bf16(pa, load_b_frag(Vp, DQ, j0, 32), o2);
    o3 = wmma_bf16(pa, load_b_frag(Vp, DQ, j0, 48), o3);
  }
#pragma unroll
  for (int v = 0; v < 8; ++v){
    float rinv = 1.0f / lrow[v];
    int r = q0 + mb + v;
    size_t rowoff = (size_t)(b * S + r) * DM + head * DQ;   // pack heads for o-proj GEMM
    O[rowoff + 0  + col] = f32_to_bf16(o0[v] * rinv);
    O[rowoff + 16 + col] = f32_to_bf16(o1[v] * rinv);
    O[rowoff + 32 + col] = f32_to_bf16(o2[v] * rinv);
    O[rowoff + 48 + col] = f32_to_bf16(o3[v] * rinv);
  }
}

// ---------- generic bf16 GEMM, 32x64 block tile, 8 waves ----------
// Double-buffered GLOBAL_LOAD_ASYNC_TO_LDS_B128 staging (ASYNCcnt).
// Per stage: waves 0-3 issue one async for the A tile (8 rows x 64B),
//            all 8 waves issue one async for the B tile (4 rows x 128B).
template<int RELU, int OUTBF16>
__global__ __launch_bounds__(256) void gemm_kernel(
    const unsigned short* __restrict__ A,
    const unsigned short* __restrict__ Bm,
    const float* __restrict__ bias,
    void* __restrict__ Cout,
    int N, int Ktot)
{
  __shared__ unsigned short As[2][32 * 32];
  __shared__ unsigned short Bs[2][32 * 64];
  int tid = threadIdx.x, wave = tid >> 5, lane = tid & 31;
  int wr = wave >> 2, wc = wave & 3;
  int row0 = blockIdx.x * 32, col0 = blockIdx.y * 64;

  // per-lane async-copy coordinates (fixed across stages)
  int brow = wave * 4 + (lane >> 3);          // B tile row 0..31
  int bcol = (lane & 7) * 8;                  // B tile col (u16), 16B chunks
  int arow = (wave & 3) * 8 + (lane >> 2);    // A tile row 0..31 (waves 0-3)
  int acol = (lane & 3) * 8;                  // A tile col (u16)

  auto issue_stage = [&](int k0, int bsel){
    async_copy_b128_to_lds(&Bm[(size_t)(k0 + brow) * N + col0 + bcol], &Bs[bsel][brow * 64 + bcol]);
    if (wave < 4)
      async_copy_b128_to_lds(&A[(size_t)(row0 + arow) * Ktot + k0 + acol], &As[bsel][arow * 32 + acol]);
  };

  v8f acc = {};
  int nk = Ktot >> 5;
  issue_stage(0, 0);
  for (int i = 0; i < nk; ++i){
    int bsel = i & 1;
    if (i + 1 < nk){
      issue_stage((i + 1) << 5, bsel ^ 1);
      // wait for stage i only (in-order ASYNCcnt; stage i+1 leaves 2 or 1 outstanding)
      if (wave < 4) asm volatile("s_wait_asynccnt 0x2" ::: "memory");
      else          asm volatile("s_wait_asynccnt 0x1" ::: "memory");
    } else {
      asm volatile("s_wait_asynccnt 0x0" ::: "memory");
    }
    __syncthreads();                           // all waves' stage-i LDS writes visible
    v16bf af = load_a_frag(As[bsel], 32, wr * 16, 0);
    v16bf bf = load_b_frag(Bs[bsel], 64, 0, wc * 16);
    acc = wmma_bf16(af, bf, acc);
    __syncthreads();                           // done reading buf before it is overwritten
  }
  int n  = col0 + wc * 16 + (lane & 15);
  int mr = row0 + wr * 16 + ((lane >> 4) << 3);
  float bv = bias ? bias[n] : 0.0f;
#pragma unroll
  for (int v = 0; v < 8; ++v){
    float xx = acc[v] + bv;
    if (RELU) xx = fmaxf(xx, 0.0f);
    if (OUTBF16) ((unsigned short*)Cout)[(size_t)(mr + v) * N + n] = f32_to_bf16(xx);
    else         ((float*)Cout)[(size_t)(mr + v) * N + n] = xx;
  }
}

// ---------- residual add + LayerNorm (one row per block) ----------
__global__ __launch_bounds__(256) void resid_ln_kernel(
    const float* __restrict__ y,
    const float* __restrict__ resid,
    const float* __restrict__ g,
    const float* __restrict__ be,
    float* __restrict__ hout,
    unsigned short* __restrict__ hbout)
{
  const int DM = 1024;
  __shared__ float sdata[256];
  int row = blockIdx.x, tid = threadIdx.x;
  size_t base = (size_t)row * DM;
  float t[4]; float s = 0.f;
#pragma unroll
  for (int i = 0; i < 4; ++i){
    int c = tid + i * 256;
    t[i] = y[base + c] + resid[base + c];
    s += t[i];
  }
  sdata[tid] = s; __syncthreads();
  for (int st = 128; st > 0; st >>= 1){ if (tid < st) sdata[tid] += sdata[tid + st]; __syncthreads(); }
  float mu = sdata[0] * (1.0f / DM);
  __syncthreads();
  float vs = 0.f;
#pragma unroll
  for (int i = 0; i < 4; ++i){ float d = t[i] - mu; vs += d * d; }
  sdata[tid] = vs; __syncthreads();
  for (int st = 128; st > 0; st >>= 1){ if (tid < st) sdata[tid] += sdata[tid + st]; __syncthreads(); }
  float rstd = rsqrtf(sdata[0] * (1.0f / DM) + 1e-5f);
#pragma unroll
  for (int i = 0; i < 4; ++i){
    int c = tid + i * 256;
    float o = (t[i] - mu) * rstd * g[c] + be[c];
    hout[base + c] = o;
    hbout[base + c] = f32_to_bf16(o);
  }
}

// ---------- host orchestration ----------
extern "C" void kernel_launch(void* const* d_in, const int* in_sizes, int n_in,
                              void* d_out, int out_size, void* d_ws, size_t ws_size,
                              hipStream_t stream){
  const int B = 8, S = 512, DM = 1024, NH = 16, DQ = 64, DFF = 4096, L = 6;
  const int BS = B * S;
  (void)in_sizes; (void)n_in; (void)out_size; (void)ws_size;
  const float* x      = (const float*)d_in[0];
  const int*   mask   = (const int*)d_in[1];
  const float* timing = (const float*)d_in[2];
  const float* wq     = (const float*)d_in[3];
  const float* wk     = (const float*)d_in[4];
  const float* wv     = (const float*)d_in[5];
  const float* wo     = (const float*)d_in[6];
  const float* ln1g   = (const float*)d_in[7];
  const float* ln1b   = (const float*)d_in[8];
  const float* ln2g   = (const float*)d_in[9];
  const float* ln2b   = (const float*)d_in[10];
  const float* w1     = (const float*)d_in[11];
  const float* b1     = (const float*)d_in[12];
  const float* w2     = (const float*)d_in[13];
  const float* b2     = (const float*)d_in[14];

  char* wsp = (char*)d_ws;
  size_t off = 0;
  auto carve = [&](size_t bytes) -> char* {
    char* p = wsp + off; off = (off + bytes + 255) & ~(size_t)255; return p;
  };
  float*          h   = (float*)         carve((size_t)BS * DM * 4);
  unsigned short* hb  = (unsigned short*)carve((size_t)BS * DM * 2);
  unsigned short* qb  = (unsigned short*)carve((size_t)BS * DM * 2);
  unsigned short* kb  = (unsigned short*)carve((size_t)BS * DM * 2);
  unsigned short* vb  = (unsigned short*)carve((size_t)BS * DM * 2);
  unsigned short* ao  = (unsigned short*)carve((size_t)BS * DM * 2);
  float*          s1  = (float*)         carve((size_t)BS * DM * 4);
  unsigned short* fa  = (unsigned short*)carve((size_t)BS * DFF * 2);
  unsigned short* wqL = (unsigned short*)carve((size_t)NH * DM * DQ * 2);
  unsigned short* wkL = (unsigned short*)carve((size_t)NH * DM * DQ * 2);
  unsigned short* wvL = (unsigned short*)carve((size_t)NH * DM * DQ * 2);
  unsigned short* woL = (unsigned short*)carve((size_t)DM * DM * 2);
  unsigned short* w1L = (unsigned short*)carve((size_t)DM * DFF * 2);
  unsigned short* w2L = (unsigned short*)carve((size_t)DFF * DM * 2);

  int total = BS * DM;
  prep_kernel<<<(total + 255) / 256, 256, 0, stream>>>(x, timing, h, hb, total);

  const int WQKV = NH * DM * DQ;   // 1,048,576 elems per layer
  const int WFF  = DM * DFF;       // 4,194,304 elems per layer
  for (int l = 0; l < L; ++l){
    cast_kernel<<<(WQKV + 255) / 256, 256, 0, stream>>>(wq + (size_t)l * WQKV, wqL, WQKV);
    cast_kernel<<<(WQKV + 255) / 256, 256, 0, stream>>>(wk + (size_t)l * WQKV, wkL, WQKV);
    cast_kernel<<<(WQKV + 255) / 256, 256, 0, stream>>>(wv + (size_t)l * WQKV, wvL, WQKV);
    cast_kernel<<<(WQKV + 255) / 256, 256, 0, stream>>>(wo + (size_t)l * WQKV, woL, WQKV);
    cast_kernel<<<(WFF  + 255) / 256, 256, 0, stream>>>(w1 + (size_t)l * WFF,  w1L, WFF);
    cast_kernel<<<(WFF  + 255) / 256, 256, 0, stream>>>(w2 + (size_t)l * WFF,  w2L, WFF);

    qkv_kernel<<<dim3(BS / 16, NH, 3), 128, 0, stream>>>(hb, wqL, wkL, wvL, qb, kb, vb);
    attn_kernel<<<dim3(S / 128, NH, B), 256, 0, stream>>>(qb, kb, vb, mask, ao);
    gemm_kernel<0,0><<<dim3(BS / 32, DM / 64), 256, 0, stream>>>(ao, woL, nullptr, s1, DM, DM);
    resid_ln_kernel<<<BS, 256, 0, stream>>>(s1, h, ln1g + (size_t)l * DM, ln1b + (size_t)l * DM, h, hb);
    gemm_kernel<1,1><<<dim3(BS / 32, DFF / 64), 256, 0, stream>>>(hb, w1L, b1 + (size_t)l * DFF, fa, DFF, DM);
    gemm_kernel<0,0><<<dim3(BS / 32, DM / 64), 256, 0, stream>>>(fa, w2L, b2 + (size_t)l * DM, s1, DM, DFF);
    float* hdst = (l == L - 1) ? (float*)d_out : h;
    resid_ln_kernel<<<BS, 256, 0, stream>>>(s1, h, ln2g + (size_t)l * DM, ln2b + (size_t)l * DM, hdst, hb);
  }
}
